// MultiHeadSelfAttention_23081154249251
// MI455X (gfx1250) — matrix operand order repaired and checked by reading, not compiled, not run on hardware
//
#include <hip/hip_runtime.h>
#include <hip/hip_bf16.h>

// MI455X (gfx1250) multi-head self-attention, f16 WMMA with f32 accumulation.
// Pipeline: cvt f32->f16, fused QKV GEMM (head-scattered f16 out), flash
// attention (online softmax, double-buffered async K/V staging to LDS),
// output-proj GEMM (f32).

typedef _Float16 v16h __attribute__((ext_vector_type(16)));
typedef _Float16 v8h  __attribute__((ext_vector_type(8)));
typedef _Float16 v4h  __attribute__((ext_vector_type(4)));
typedef float    v8f  __attribute__((ext_vector_type(8)));
typedef int      i32x4 __attribute__((vector_size(16)));

#define B_  4
#define T_  2048
#define D_  1024
#define H_  16
#define HD_ 64
#define BT_ 8192   // B_*T_

// ---------------------------------------------------------------------------
// WMMA helpers (CDNA5 16x16x32 f16 -> f32)
// ---------------------------------------------------------------------------
__device__ __forceinline__ v8f wmma16(v16h a, v16h b, v8f c) {
  // (neg_a, A, neg_b, B, c_mod, C, reuse_a, reuse_b)
  return __builtin_amdgcn_wmma_f32_16x16x32_f16(false, a, false, b, (short)0, c,
                                                false, false);
}

// A-matrix fragment (16x32 f16, row-major source):
// lane m = lane&15 holds row m; half-wave selects K-pattern:
//   group 0: K = 0..15, group 1: K = 16..31
// kb = (lane>>4)*8 (offset 2*kb).  Two 16B loads per fragment.
__device__ __forceinline__ v16h load_fragA(const _Float16* rowp, int kb) {
  v8h lo = *(const v8h*)(rowp + 2 * kb);
  v8h hi = *(const v8h*)(rowp + 2 * kb + 8);
  return __builtin_shufflevector(lo, hi, 0, 1, 2, 3, 4, 5, 6, 7, 8, 9, 10, 11,
                                 12, 13, 14, 15);
}

// B-matrix fragment (32x16 f16), source stored K-contiguous per column
// (column n of B == row n of a (N,K) row-major matrix):
// lane n = lane&15 holds column n; half-wave g holds K = g*16 .. g*16+15.
// g16 = (lane>>4)*16.  Two 16B loads per fragment.
__device__ __forceinline__ v16h load_fragB(const _Float16* colp, int g16) {
  v8h lo = *(const v8h*)(colp + g16);
  v8h hi = *(const v8h*)(colp + g16 + 8);
  return __builtin_shufflevector(lo, hi, 0, 1, 2, 3, 4, 5, 6, 7, 8, 9, 10, 11,
                                 12, 13, 14, 15);
}

// ---------------------------------------------------------------------------
// Async global -> LDS copy of 16 bytes (gfx1250 GLOBAL_LOAD_ASYNC_TO_LDS_B128,
// tracked by ASYNCcnt).  Builtin signature (from compiler diagnostic):
//   (i32x4 __device__*, i32x4 __shared__*, imm offset, imm cpol)
// ---------------------------------------------------------------------------
__device__ __forceinline__ void async_ld16(const _Float16* gsrc,
                                           _Float16* ldst) {
#if __has_builtin(__builtin_amdgcn_global_load_async_to_lds_b128)
  __builtin_amdgcn_global_load_async_to_lds_b128(
      (__attribute__((address_space(1))) i32x4*)gsrc,
      (__attribute__((address_space(3))) i32x4*)ldst, 0, 0);
#else
  // ISA 10.2: low 32 bits of a flat LDS address are the LDS byte offset.
  unsigned loff = (unsigned)(size_t)ldst;
  unsigned long long ga = (unsigned long long)(size_t)gsrc;
  asm volatile("global_load_async_to_lds_b128 %0, %1, off"
               :
               : "v"(loff), "v"(ga)
               : "memory");
#endif
}

#if __has_builtin(__builtin_amdgcn_s_wait_asynccnt)
#define WAIT_ASYNC(n) __builtin_amdgcn_s_wait_asynccnt(n)
#else
#define WAIT_ASYNC(n) asm volatile("s_wait_asynccnt %0" ::"n"(n) : "memory")
#endif

// ---------------------------------------------------------------------------
// f32 -> f16 conversion (4 elems/thread, vectorized)
// ---------------------------------------------------------------------------
__global__ __launch_bounds__(256) void cvt_f32_f16(const float* __restrict__ src,
                                                   _Float16* __restrict__ dst,
                                                   int n) {
  int i = (blockIdx.x * 256 + threadIdx.x) * 4;
  if (i < n) {
    float4 f = *(const float4*)(src + i);
    v4h h = {(_Float16)f.x, (_Float16)f.y, (_Float16)f.z, (_Float16)f.w};
    *(v4h*)(dst + i) = h;
  }
}

// ---------------------------------------------------------------------------
// Shared GEMM block:  C(128x128) = A(128x1024) * W^T  where W is (N,K) f16
// row-major (torch Linear convention), i.e. B[k][n] = W[n][k].
// 256 threads = 8 waves in a 2(M) x 4(N) wave grid; each wave owns 4x2
// WMMA tiles (64x32).  A and W tiles staged through LDS (16 KB).
// ---------------------------------------------------------------------------
__device__ __forceinline__ void gemm_block(const _Float16* __restrict__ A,
                                           const _Float16* __restrict__ Wn,
                                           int rowBase, int colBase,
                                           _Float16* sA, _Float16* sB, int tid,
                                           v8f (&acc)[4][2]) {
  const int lane = tid & 31;
  const int wave = tid >> 5;
  const int wm = wave >> 2;          // 0..1  (64 rows each)
  const int wn = wave & 3;           // 0..3  (32 cols each)
  const int lm = lane & 15;
  const int kb = (lane >> 4) * 8;    // A-fragment K pattern
  const int g16 = (lane >> 4) * 16;  // B-fragment K pattern
  const int cprow = tid >> 1;        // 0..127
  const int cpcol = (tid & 1) * 16;  // 0 or 16

  for (int k0 = 0; k0 < D_; k0 += 32) {
    const _Float16* arow = A + (size_t)(rowBase + cprow) * D_ + k0 + cpcol;
    const _Float16* brow = Wn + (size_t)(colBase + cprow) * D_ + k0 + cpcol;
    *(v8h*)(sA + cprow * 32 + cpcol) = *(const v8h*)(arow);
    *(v8h*)(sA + cprow * 32 + cpcol + 8) = *(const v8h*)(arow + 8);
    *(v8h*)(sB + cprow * 32 + cpcol) = *(const v8h*)(brow);
    *(v8h*)(sB + cprow * 32 + cpcol + 8) = *(const v8h*)(brow + 8);
    if (k0 + 32 < D_) {
      __builtin_prefetch(arow + 32, 0, 0);
      __builtin_prefetch(brow + 32, 0, 0);
    }
    __syncthreads();

    v16h bf0 = load_fragB(sB + (wn * 32 + lm) * 32, g16);
    v16h bf1 = load_fragB(sB + (wn * 32 + 16 + lm) * 32, g16);
#pragma unroll
    for (int mt = 0; mt < 4; mt++) {
      v16h af = load_fragA(sA + (wm * 64 + mt * 16 + lm) * 32, kb);
      acc[mt][0] = wmma16(af, bf0, acc[mt][0]);
      acc[mt][1] = wmma16(af, bf1, acc[mt][1]);
    }
    __syncthreads();
  }
}

// ---------------------------------------------------------------------------
// Fused QKV projection: z in {0,1,2} selects Q/K/V.  Output is f16 in
// (B*H, T, HD) layout for the attention kernel.
// ---------------------------------------------------------------------------
__global__ __launch_bounds__(256) void qkv_proj_kernel(
    const _Float16* __restrict__ xh, const _Float16* __restrict__ wh,
    const float* __restrict__ bq, const float* __restrict__ bk,
    const float* __restrict__ bv, _Float16* __restrict__ qh,
    _Float16* __restrict__ kh, _Float16* __restrict__ vh) {
  __shared__ _Float16 sA[128 * 32];
  __shared__ _Float16 sB[128 * 32];

  const int z = blockIdx.z;
  const _Float16* W = wh + (size_t)z * D_ * D_;
  const float* bias = (z == 0) ? bq : (z == 1) ? bk : bv;
  _Float16* outp = (z == 0) ? qh : (z == 1) ? kh : vh;

  const int tid = threadIdx.x;
  const int lane = tid & 31;
  const int wave = tid >> 5;
  const int wm = wave >> 2, wn = wave & 3;
  const int lm = lane & 15;
  const int rowBase = blockIdx.y * 128;
  const int colBase = blockIdx.x * 128;

  v8f acc[4][2] = {};
  gemm_block(xh, W, rowBase, colBase, sA, sB, tid, acc);

#pragma unroll
  for (int mt = 0; mt < 4; mt++) {
#pragma unroll
    for (int nt = 0; nt < 2; nt++) {
      const int o = colBase + wn * 32 + nt * 16 + lm;  // output feature
      const float bval = bias[o];
      const int h = o >> 6;
      const int hd = o & 63;
#pragma unroll
      for (int r = 0; r < 8; r++) {
        const int m = rowBase + wm * 64 + mt * 16 + (lane >> 4) * 8 + r;
        const int b = m >> 11;      // / T_
        const int t = m & (T_ - 1);
        outp[(((size_t)(b * H_ + h)) * T_ + t) * HD_ + hd] =
            (_Float16)(acc[mt][nt][r] + bval);
      }
    }
  }
}

// ---------------------------------------------------------------------------
// Flash attention: block = 128 queries of one (b,h); 8 waves x 16 rows.
// Keys streamed in chunks of 32, double-buffered in LDS via async
// global->LDS loads (ASYNCcnt).  Online softmax with 16-lane shuffle
// reductions (C-layout rows span a half-wave).
// ---------------------------------------------------------------------------
__global__ __launch_bounds__(256) void attn_kernel(
    const _Float16* __restrict__ qh, const _Float16* __restrict__ kh,
    const _Float16* __restrict__ vh, const int* __restrict__ mask,
    _Float16* __restrict__ ctx) {
  __shared__ _Float16 sK[2][32 * 64];
  __shared__ _Float16 sV[2][32 * 64];
  __shared__ _Float16 sP[8][16 * 32];

  const int bh = blockIdx.y;   // b*H + h
  const int b = bh >> 4;
  const int h = bh & (H_ - 1);
  const int tid = threadIdx.x;
  const int lane = tid & 31;
  const int wave = tid >> 5;
  const int lm = lane & 15;
  const int kb = (lane >> 4) * 8;
  const int g16 = (lane >> 4) * 16;
  const int qRow0 = blockIdx.x * 128 + wave * 16;

  const _Float16* Qb = qh + (size_t)bh * T_ * HD_;
  const _Float16* Kb = kh + (size_t)bh * T_ * HD_;
  const _Float16* Vb = vh + (size_t)bh * T_ * HD_;

  // per-thread slice of the cooperative K/V chunk staging (16B each)
  const int srow = tid >> 3;           // 0..31
  const int scol = (tid & 7) * 8;      // 0..56

  // Q fragments: 16 rows x HD=64, split into two K-chunks of 32.
  v16h qf[2];
#pragma unroll
  for (int c = 0; c < 2; c++)
    qf[c] = load_fragA(Qb + (size_t)(qRow0 + lm) * HD_ + c * 32, kb);

  float rmax[8], rsum[8];
#pragma unroll
  for (int r = 0; r < 8; r++) { rmax[r] = -3.0e38f; rsum[r] = 0.0f; }
  v8f oacc[4] = {};  // 16 x 64 f32 output accumulator (4 N-tiles)

  const float scale = 0.125f;  // 1/sqrt(HD)

  // prime buffer 0 with the first key chunk (2 async b128 per thread)
  async_ld16(Kb + (size_t)srow * HD_ + scol, &sK[0][srow * 64 + scol]);
  async_ld16(Vb + (size_t)srow * HD_ + scol, &sV[0][srow * 64 + scol]);

  for (int n0 = 0; n0 < T_; n0 += 32) {
    const int cur = (n0 >> 5) & 1;
    const bool pre = (n0 + 32) < T_;
    if (pre) {  // stream next chunk into the other buffer
      async_ld16(Kb + (size_t)(n0 + 32 + srow) * HD_ + scol,
                 &sK[cur ^ 1][srow * 64 + scol]);
      async_ld16(Vb + (size_t)(n0 + 32 + srow) * HD_ + scol,
                 &sV[cur ^ 1][srow * 64 + scol]);
      WAIT_ASYNC(2);  // async loads retire in order: chunk-cur data landed
    } else {
      WAIT_ASYNC(0);
    }
    __syncthreads();
    const _Float16* cK = sK[cur];
    const _Float16* cV = sV[cur];

    // S = Q * K^T for 32 keys as two 16x16 tiles
    v8f s[2];
#pragma unroll
    for (int j = 0; j < 2; j++) {
      v8f cacc = {};
#pragma unroll
      for (int c = 0; c < 2; c++) {
        v16h kf = load_fragB(cK + (j * 16 + lm) * 64 + c * 32, g16);
        cacc = wmma16(qf[c], kf, cacc);
      }
      const int key = n0 + j * 16 + lm;   // this lane's key column
      const bool mok = mask[b * T_ + key] != 0;
#pragma unroll
      for (int r = 0; r < 8; r++) {
        const float v = cacc[r] * scale;
        cacc[r] = mok ? v : -3.0e38f;
      }
      s[j] = cacc;
    }

    // online softmax: per-row max/sum, rows span a 16-lane half-wave
    float alpha[8];
#pragma unroll
    for (int r = 0; r < 8; r++) {
      float m = fmaxf(s[0][r], s[1][r]);
#pragma unroll
      for (int x = 1; x < 16; x <<= 1) m = fmaxf(m, __shfl_xor(m, x, 32));
      const float nm = fmaxf(rmax[r], m);
      alpha[r] = __expf(rmax[r] - nm);
      rmax[r] = nm;
      const float p0 = __expf(s[0][r] - nm);
      const float p1 = __expf(s[1][r] - nm);
      s[0][r] = p0;
      s[1][r] = p1;
      float ps = p0 + p1;
#pragma unroll
      for (int x = 1; x < 16; x <<= 1) ps += __shfl_xor(ps, x, 32);
      rsum[r] = rsum[r] * alpha[r] + ps;
    }

    // rescale running output
#pragma unroll
    for (int vt = 0; vt < 4; vt++)
#pragma unroll
      for (int r = 0; r < 8; r++) oacc[vt][r] = oacc[vt][r] * alpha[r];

    // P: C-layout -> A-layout via this wave's private LDS slab
    {
      _Float16* ph = sP[wave];
      const int mrow = (lane >> 4) * 8;
#pragma unroll
      for (int j = 0; j < 2; j++)
#pragma unroll
        for (int r = 0; r < 8; r++)
          ph[(mrow + r) * 32 + j * 16 + lm] = (_Float16)s[j][r];
    }
    asm volatile("s_wait_dscnt 0" ::: "memory");  // cross-lane LDS visibility

    v16h pf = load_fragA(sP[wave] + lm * 32, kb);

    // O += P * V   (four 16-wide HD tiles)
#pragma unroll
    for (int vt = 0; vt < 4; vt++) {
      v16h vf;
#pragma unroll
      for (int i = 0; i < 16; i++)
        vf[i] = cV[(g16 + i) * 64 + vt * 16 + lm];
      oacc[vt] = wmma16(pf, vf, oacc[vt]);
    }
    __syncthreads();
  }

  // normalize and scatter back to (B, T, D) f16 context
  float rinv[8];
#pragma unroll
  for (int r = 0; r < 8; r++) rinv[r] = 1.0f / rsum[r];
#pragma unroll
  for (int vt = 0; vt < 4; vt++) {
    const int hd = vt * 16 + lm;
#pragma unroll
    for (int r = 0; r < 8; r++) {
      const int q = qRow0 + (lane >> 4) * 8 + r;
      ctx[((size_t)(b * T_ + q)) * D_ + h * HD_ + hd] =
          (_Float16)(oacc[vt][r] * rinv[r]);
    }
  }
}

// ---------------------------------------------------------------------------
// Output projection: out = ctx @ Wo^T + bo   (f32 output)
// ---------------------------------------------------------------------------
__global__ __launch_bounds__(256) void out_proj_kernel(
    const _Float16* __restrict__ ah, const _Float16* __restrict__ W,
    const float* __restrict__ bias, float* __restrict__ out) {
  __shared__ _Float16 sA[128 * 32];
  __shared__ _Float16 sB[128 * 32];

  const int tid = threadIdx.x;
  const int lane = tid & 31;
  const int wave = tid >> 5;
  const int wm = wave >> 2, wn = wave & 3;
  const int lm = lane & 15;
  const int rowBase = blockIdx.y * 128;
  const int colBase = blockIdx.x * 128;

  v8f acc[4][2] = {};
  gemm_block(ah, W, rowBase, colBase, sA, sB, tid, acc);

#pragma unroll
  for (int mt = 0; mt < 4; mt++) {
#pragma unroll
    for (int nt = 0; nt < 2; nt++) {
      const int o = colBase + wn * 32 + nt * 16 + lm;
      const float bval = bias[o];
#pragma unroll
      for (int r = 0; r < 8; r++) {
        const int m = rowBase + wm * 64 + mt * 16 + (lane >> 4) * 8 + r;
        out[(size_t)m * D_ + o] = acc[mt][nt][r] + bval;
      }
    }
  }
}

// ---------------------------------------------------------------------------
// Host entry.  Workspace layout (all f16, ~92 MB total):
//   x_h[8192x1024] | w_h[4x1024x1024] (q,k,v,o) | q_h | k_h | v_h | ctx_h
// ---------------------------------------------------------------------------
extern "C" void kernel_launch(void* const* d_in, const int* in_sizes, int n_in,
                              void* d_out, int out_size, void* d_ws,
                              size_t ws_size, hipStream_t stream) {
  (void)in_sizes; (void)n_in; (void)out_size; (void)ws_size;

  const float* x = (const float*)d_in[0];
  const float* Wq = (const float*)d_in[1];
  const float* bq = (const float*)d_in[2];
  const float* Wk = (const float*)d_in[3];
  const float* bk = (const float*)d_in[4];
  const float* Wv = (const float*)d_in[5];
  const float* bv = (const float*)d_in[6];
  const float* Wo = (const float*)d_in[7];
  const float* bo = (const float*)d_in[8];
  const int* mask = (const int*)d_in[9];
  float* out = (float*)d_out;

  char* p = (char*)d_ws;
  _Float16* x_h = (_Float16*)p; p += (size_t)BT_ * D_ * 2;
  _Float16* w_h = (_Float16*)p; p += (size_t)4 * D_ * D_ * 2;
  _Float16* q_h = (_Float16*)p; p += (size_t)BT_ * D_ * 2;
  _Float16* k_h = (_Float16*)p; p += (size_t)BT_ * D_ * 2;
  _Float16* v_h = (_Float16*)p; p += (size_t)BT_ * D_ * 2;
  _Float16* c_h = (_Float16*)p; p += (size_t)BT_ * D_ * 2;

  const int nx = BT_ * D_;
  const int nw = D_ * D_;
  cvt_f32_f16<<<nx / 4 / 256, 256, 0, stream>>>(x, x_h, nx);
  cvt_f32_f16<<<nw / 4 / 256, 256, 0, stream>>>(Wq, w_h + 0 * (size_t)nw, nw);
  cvt_f32_f16<<<nw / 4 / 256, 256, 0, stream>>>(Wk, w_h + 1 * (size_t)nw, nw);
  cvt_f32_f16<<<nw / 4 / 256, 256, 0, stream>>>(Wv, w_h + 2 * (size_t)nw, nw);
  cvt_f32_f16<<<nw / 4 / 256, 256, 0, stream>>>(Wo, w_h + 3 * (size_t)nw, nw);

  qkv_proj_kernel<<<dim3(D_ / 128, BT_ / 128, 3), 256, 0, stream>>>(
      x_h, w_h, bq, bk, bv, q_h, k_h, v_h);

  attn_kernel<<<dim3(T_ / 128, B_ * H_), 256, 0, stream>>>(q_h, k_h, v_h, mask,
                                                           c_h);

  out_proj_kernel<<<dim3(D_ / 128, BT_ / 128), 256, 0, stream>>>(
      c_h, w_h + 3 * (size_t)nw, bo, out);
}
